// LastAttenion_833223655696
// MI455X (gfx1250) — matrix operand order repaired and checked by reading, not compiled, not run on hardware
//
#include <hip/hip_runtime.h>
#include <math.h>

// Problem constants (from reference)
constexpr int NB = 512;    // batch
constexpr int NL = 200;    // sequence length
constexpr int NH = 512;    // hidden size
constexpr int NHEADS = 8;  // heads
constexpr int NLP = 224;   // L padded to multiple of 32 (K-dim for wsum pass)

typedef __attribute__((ext_vector_type(16))) _Float16 v16h;
typedef __attribute__((ext_vector_type(8)))  _Float16 v8h;
typedef __attribute__((ext_vector_type(8)))  float    v8f;
typedef __attribute__((ext_vector_type(4)))  float    v4f;
typedef __attribute__((ext_vector_type(4)))  unsigned int u32x4;
typedef __attribute__((ext_vector_type(8)))  int      i32x8;
typedef __attribute__((ext_vector_type(4)))  int      i32x4;

#ifndef __has_builtin
#define __has_builtin(x) 0
#endif
#if __has_builtin(__builtin_amdgcn_tensor_load_to_lds) && \
    __has_builtin(__builtin_amdgcn_s_wait_tensorcnt)
#define HAS_TDM 1
#else
#define HAS_TDM 0
#endif

// ---------------------------------------------------------------------------
// Tensor Data Mover: 1-D contiguous DMA global -> LDS (ndw dwords).
// D# per CDNA5 ISA: group0 = {count=1, lds_addr, global_addr,
// type=2}; group1 = {data_size=4B, tensor_dim0=tile_dim0=ndw, dims1=1,
// stride0=ndw}; groups 2/3 zero (<=2D tensor). This toolchain's builtin takes
// the 6-arg form (extra i32x8 group, zero for <=2D) + cpol.
// ---------------------------------------------------------------------------
#if HAS_TDM
__device__ __forceinline__ void tdm_load_1d(unsigned lds_off, const void* gsrc,
                                            unsigned ndw) {
  unsigned long long ga = (unsigned long long)(size_t)gsrc;
  u32x4 g0;
  g0[0] = 1u;                                               // count=1, user mode
  g0[1] = lds_off;                                          // LDS byte address
  g0[2] = (unsigned)ga;                                     // global_addr[31:0]
  g0[3] = (unsigned)((ga >> 32) & 0x01ffffffu) | (2u << 30);// addr[56:32] | type=2
  i32x8 g1;
  g1[0] = (int)(2u << 16);                          // data_size=2 -> 4 bytes
  g1[1] = (int)((ndw & 0xffffu) << 16);             // tensor_dim0[15:0]
  g1[2] = (int)(((ndw >> 16) & 0xffffu) | (1u << 16)); // dim0[31:16], tensor_dim1=1
  g1[3] = (int)((ndw & 0xffffu) << 16);             // tile_dim0
  g1[4] = 1;                                        // tile_dim1=1, tile_dim2=0
  g1[5] = (int)ndw;                                 // tensor_dim0_stride[31:0]
  g1[6] = 0;
  g1[7] = 0;
  i32x4 z4 = {0, 0, 0, 0};
  i32x8 z8 = {0, 0, 0, 0, 0, 0, 0, 0};
  __builtin_amdgcn_tensor_load_to_lds(g0, g1, z4, z4, z8, 0);
}
#endif

// ---------------------------------------------------------------------------
// WMMA fragment helpers (CDNA5 gfx1250, wave32). Layouts used by these helpers:
//  A (16x32 f16): lane m = lane%16, sel = lane/16.
//    element e: K = sel*16 + e (same order as B)
//  B (32x16 f16): lane n = lane%16, sel = lane/16; element e: K = sel*16 + e
//  C/D (16x16 f32): vgpr v: M = v + 8*(lane>=16), N = lane%16
// All contiguous chunks are 16B-aligned -> b128 loads.
// ---------------------------------------------------------------------------
__device__ __forceinline__ v8f wmma16(v16h a, v16h b, v8f c) {
  return __builtin_amdgcn_wmma_f32_16x16x32_f16(
      /*neg_a=*/false, a, /*neg_b=*/false, b,
      /*c_mod=*/(short)0, c, /*reuse_a=*/false, /*reuse_b=*/false);
}

__device__ __forceinline__ v16h cvt16(v4f c0, v4f c1, v4f c2, v4f c3) {
  v16h r;
#pragma unroll
  for (int t = 0; t < 4; ++t) {
    r[t]      = (_Float16)c0[t];
    r[4 + t]  = (_Float16)c1[t];
    r[8 + t]  = (_Float16)c2[t];
    r[12 + t] = (_Float16)c3[t];
  }
  return r;
}

// A fragment from f32 row (two 32B chunks -> 4x b128)
__device__ __forceinline__ v16h ldA_f32(const float* row, int sel) {
  const v4f* p0 = (const v4f*)(row + sel * 16);
  const v4f* p1 = (const v4f*)(row + sel * 16 + 8);
  return cvt16(p0[0], p0[1], p1[0], p1[1]);
}

// B fragment, contiguous K (64B chunk -> 4x b128)
__device__ __forceinline__ v16h ldB_f32c(const float* base, int sel, bool valid) {
  v16h b;
  if (valid) {
    const v4f* p = (const v4f*)(base + sel * 16);
    b = cvt16(p[0], p[1], p[2], p[3]);
  } else {
#pragma unroll
    for (int t = 0; t < 16; ++t) b[t] = (_Float16)0.0f;
  }
  return b;
}

// B fragment, strided K (column access)
__device__ __forceinline__ v16h ldB_f32s(const float* base, int sel, int strideK) {
  v16h b;
#pragma unroll
  for (int t = 0; t < 16; ++t) b[t] = (_Float16)base[(sel * 16 + t) * (long)strideK];
  return b;
}

// A fragment from f16 LDS row (two 16B chunks -> ds_load_b128)
__device__ __forceinline__ v16h ldA_h16(const _Float16* row, int sel, bool valid) {
  v16h a;
  if (valid) {
    v8h c0 = *(const v8h*)(row + sel * 16);
    v8h c1 = *(const v8h*)(row + sel * 16 + 8);
#pragma unroll
    for (int t = 0; t < 8; ++t) { a[t] = c0[t]; a[8 + t] = c1[t]; }
  } else {
#pragma unroll
    for (int t = 0; t < 16; ++t) a[t] = (_Float16)0.0f;
  }
  return a;
}

// ---------------------------------------------------------------------------
// K1: q0 = ht1 @ W0^T + b0   (M=512 batch, N=512, K=512)
// grid (32 ntile, 32 mtile), block 32 (one wave per 16x16 tile)
// ---------------------------------------------------------------------------
__global__ void k_q0(const float* __restrict__ ht1, const float* __restrict__ W0,
                     const float* __restrict__ b0, float* __restrict__ q0) {
  const int lane = threadIdx.x, sel = lane >> 4, idx = lane & 15;
  const int n0 = blockIdx.x * 16, m0 = blockIdx.y * 16;
  const float* arow = ht1 + (long)(m0 + idx) * NH;   // A[m][k] = ht1[m0+m][k]
  const float* brow = W0 + (long)(n0 + idx) * NH;    // B[k][n] = W0[n0+n][k]
  v8f acc = {};
  for (int k0 = 0; k0 < NH; k0 += 32) {
    v16h a = ldA_f32(arow + k0, sel);
    v16h b = ldB_f32c(brow + k0, sel, true);
    acc = wmma16(a, b, acc);
  }
#pragma unroll
  for (int v = 0; v < 8; ++v) {
    int m = v + 8 * sel;
    q0[(long)(m0 + m) * NH + n0 + idx] = acc[v] + b0[n0 + idx];
  }
}

// ---------------------------------------------------------------------------
// K2: c[b,h] = q0[b, h*64 : h*64+64] . b1[h*64 : h*64+64]
// ---------------------------------------------------------------------------
__global__ void k_c(const float* __restrict__ q0, const float* __restrict__ b1,
                    float* __restrict__ c) {
  const int b = blockIdx.x, h = threadIdx.x >> 5, lane = threadIdx.x & 31;
  const float* qr = q0 + (long)b * NH + h * 64;
  const float* br = b1 + h * 64;
  float s = qr[lane] * br[lane] + qr[lane + 32] * br[lane + 32];
  for (int o = 16; o > 0; o >>= 1) s += __shfl_xor(s, o, 32);
  if (lane == 0) c[b * NHEADS + h] = s;
}

// ---------------------------------------------------------------------------
// K3: u16[b,h,j] = sum_{d<64} q0[b, h*64+d] * W1[h*64+d, j]   (f16 output)
// Per head: GEMM M=512(b), N=512(j), K=64.  grid (32 ntile, 32 mtile, 8 h)
// ---------------------------------------------------------------------------
__global__ void k_u(const float* __restrict__ q0, const float* __restrict__ W1,
                    _Float16* __restrict__ u16) {
  const int lane = threadIdx.x, sel = lane >> 4, idx = lane & 15;
  const int n0 = blockIdx.x * 16, m0 = blockIdx.y * 16, h = blockIdx.z;
  const float* arow = q0 + (long)(m0 + idx) * NH + h * 64;   // A[m][k]=q0[b,h*64+k]
  const float* bbase = W1 + (long)h * 64 * NH + n0 + idx;    // B[k][n]=W1[h*64+k][n0+n]
  v8f acc = {};
  for (int k0 = 0; k0 < 64; k0 += 32) {
    v16h a = ldA_f32(arow + k0, sel);
    v16h b = ldB_f32s(bbase + (long)k0 * NH, sel, NH);
    acc = wmma16(a, b, acc);
  }
#pragma unroll
  for (int v = 0; v < 8; ++v) {
    int m = v + 8 * sel;
    u16[((long)(m0 + m) * NHEADS + h) * NH + n0 + idx] = (_Float16)acc[v];
  }
}

// ---------------------------------------------------------------------------
// K4: score pass (streams `hidden` once).  Per b: (8x200) = U(8x512) x hid^T
// sig[b,l,h] = sigmoid(score + c[b,h]).
// grid NB, block 416 = 13 waves (one 16-wide L tile each). u[b] staged in LDS
// via TDM (fallback: vectorized cooperative copy).
// ---------------------------------------------------------------------------
__global__ void k_scores(const _Float16* __restrict__ u16,
                         const float* __restrict__ hidden,
                         const float* __restrict__ c,
                         float* __restrict__ sig) {
  __shared__ __align__(16) _Float16 uS[NHEADS * NH];  // 8 KB
  const int b = blockIdx.x;
  const long ubase = (long)b * NHEADS * NH;
#if HAS_TDM
  if (threadIdx.x < 32) {  // one wave issues the DMA, waits on TENSORcnt
    tdm_load_1d((unsigned)(size_t)(void*)uS, u16 + ubase, (NHEADS * NH * 2) / 4);
    __builtin_amdgcn_s_wait_tensorcnt(0);
  }
  __syncthreads();
#else
  {
    const u32x4* src = (const u32x4*)(u16 + ubase);
    u32x4* dst = (u32x4*)uS;
    for (int i = threadIdx.x; i < (NHEADS * NH) / 8; i += blockDim.x) dst[i] = src[i];
  }
  __syncthreads();
#endif

  const int wave = threadIdx.x >> 5;
  const int lane = threadIdx.x & 31, sel = lane >> 4, idx = lane & 15;
  const int l = wave * 16 + idx;          // this lane's B-column (sequence pos)
  const bool lvalid = (l < NL);
  const bool avalid = (idx < 8);          // heads 0..7 valid, 8..15 zero-padded
  const _Float16* arow = uS + idx * NH;
  const float* brow = hidden + ((long)b * NL + (lvalid ? l : 0)) * NH;

  v8f acc = {};
  for (int k0 = 0; k0 < NH; k0 += 32) {
    __builtin_prefetch(brow + k0 + 128, 0, 0);
    v16h a = ldA_h16(arow + k0, sel, avalid);
    v16h bb = ldB_f32c(brow + k0, sel, lvalid);
    acc = wmma16(a, bb, acc);
  }
  if (sel == 0 && lvalid) {               // lanes 0-15 hold M=v (heads 0..7)
    v4f s0, s1;
#pragma unroll
    for (int v = 0; v < 4; ++v) {
      float x0 = acc[v] + c[b * NHEADS + v];
      float x1 = acc[v + 4] + c[b * NHEADS + v + 4];
      s0[v] = 1.0f / (1.0f + __expf(-x0));
      s1[v] = 1.0f / (1.0f + __expf(-x1));
    }
    v4f* dst = (v4f*)(sig + ((long)b * NL + l) * NHEADS);
    dst[0] = s0;
    dst[1] = s1;
  }
}

// ---------------------------------------------------------------------------
// K5: softmax over L of 2*sig; emits alpha (output #2), masked f16 weights
// (L padded to 224 with zeros), and S[b,h] = sum_l alpha*mask.
// grid NB, block 256 (wave w == head w).
// ---------------------------------------------------------------------------
__global__ void k_softmax(const float* __restrict__ sig, const int* __restrict__ mask,
                          float* __restrict__ alphaOut, _Float16* __restrict__ wgt,
                          float* __restrict__ S) {
  const int b = blockIdx.x, h = threadIdx.x >> 5, lane = threadIdx.x & 31;
  float xs[7];
  float mx = -1e30f;
#pragma unroll
  for (int i = 0; i < 7; ++i) {
    int l = i * 32 + lane;
    float x = (l < NL) ? sig[((long)b * NL + l) * NHEADS + h] : -1e30f;
    xs[i] = x;
    mx = fmaxf(mx, x);
  }
  for (int o = 16; o > 0; o >>= 1) mx = fmaxf(mx, __shfl_xor(mx, o, 32));
  float es[7];
  float sum = 0.0f;
#pragma unroll
  for (int i = 0; i < 7; ++i) {
    int l = i * 32 + lane;
    float e = (l < NL) ? __expf(2.0f * (xs[i] - mx)) : 0.0f;
    es[i] = e;
    sum += e;
  }
  for (int o = 16; o > 0; o >>= 1) sum += __shfl_xor(sum, o, 32);
  const float inv = 1.0f / sum;
  float sm = 0.0f;
#pragma unroll
  for (int i = 0; i < 7; ++i) {
    int l = i * 32 + lane;
    if (l < NL) {
      float al = es[i] * inv;
      alphaOut[((long)b * NL + l) * NHEADS + h] = al;
      float w = al * (float)mask[b * NL + l];
      wgt[((long)b * NHEADS + h) * NLP + l] = (_Float16)w;
      sm += w;
    } else {
      wgt[((long)b * NHEADS + h) * NLP + l] = (_Float16)0.0f;  // K-pad
    }
  }
  for (int o = 16; o > 0; o >>= 1) sm += __shfl_xor(sm, o, 32);
  if (lane == 0) S[b * NHEADS + h] = sm;
}

// ---------------------------------------------------------------------------
// K6: weighted-sum pass (streams `hidden` again; mostly L2 hits from pass 1).
// G[b,h,j] = sum_l wgt[b,h,l] * hidden[b,l,j].  Per b: (16x224) x (224x512).
// grid (4 jgroup, NB), block 256 (8 waves, one 16-col j tile each).
// ---------------------------------------------------------------------------
__global__ void k_wsum(const _Float16* __restrict__ wgt,
                       const float* __restrict__ hidden,
                       float* __restrict__ G) {
  __shared__ __align__(16) _Float16 wS[NHEADS * NLP];  // 3.5 KB
  const int b = blockIdx.y;
  const long wbase = (long)b * NHEADS * NLP;
#if HAS_TDM
  if (threadIdx.x < 32) {
    tdm_load_1d((unsigned)(size_t)(void*)wS, wgt + wbase, (NHEADS * NLP * 2) / 4);
    __builtin_amdgcn_s_wait_tensorcnt(0);
  }
  __syncthreads();
#else
  {
    const u32x4* src = (const u32x4*)(wgt + wbase);
    u32x4* dst = (u32x4*)wS;
    for (int i = threadIdx.x; i < (NHEADS * NLP) / 8; i += blockDim.x) dst[i] = src[i];
  }
  __syncthreads();
#endif

  const int wave = threadIdx.x >> 5;
  const int lane = threadIdx.x & 31, sel = lane >> 4, idx = lane & 15;
  const int j0 = (blockIdx.x * 8 + wave) * 16;
  const bool avalid = (idx < 8);
  const _Float16* arow = wS + idx * NLP;
  const float* hb = hidden + (long)b * NL * NH + j0 + idx;  // column slice base

  v8f acc = {};
  for (int k0 = 0; k0 < NLP; k0 += 32) {
    v16h a = ldA_h16(arow + k0, sel, avalid);
    v16h bb;  // B[k][n] = hidden[b, l=k0+k, j0+n]; clamp rows >= NL (A is 0 there)
#pragma unroll
    for (int t = 0; t < 16; ++t) {
      int r = k0 + sel * 16 + t;
      r = (r < NL) ? r : (NL - 1);
      bb[t] = (_Float16)hb[(long)r * NH];
    }
    acc = wmma16(a, bb, acc);
  }
  if (sel == 0) {
#pragma unroll
    for (int v = 0; v < 8; ++v)
      G[((long)b * NHEADS + v) * NH + j0 + idx] = acc[v];
  }
}

// ---------------------------------------------------------------------------
// K7: per-head projection: aPre[b, h*64+d] = W2_h row . G[b,h,:] + b2*S[b,h]
// Per head: GEMM M=512(b), N=64(d), K=512(j). grid (4 ntile, 32 mtile, 8 h).
// ---------------------------------------------------------------------------
__global__ void k_proj(const float* __restrict__ G, const float* __restrict__ W2,
                       const float* __restrict__ b2, const float* __restrict__ S,
                       float* __restrict__ aPre) {
  const int lane = threadIdx.x, sel = lane >> 4, idx = lane & 15;
  const int n0 = blockIdx.x * 16, m0 = blockIdx.y * 16, h = blockIdx.z;
  const float* arow = G + ((long)(m0 + idx) * NHEADS + h) * NH;  // A[m][k]=G[b,h,k]
  const float* brow = W2 + (long)(h * 64 + n0 + idx) * NH;       // B[k][n]=W2[h*64+n][k]
  v8f acc = {};
  for (int k0 = 0; k0 < NH; k0 += 32) {
    v16h a = ldA_f32(arow + k0, sel);
    v16h b = ldB_f32c(brow + k0, sel, true);
    acc = wmma16(a, b, acc);
  }
  const int col = h * 64 + n0 + idx;
#pragma unroll
  for (int v = 0; v < 8; ++v) {
    int bi = m0 + v + 8 * sel;
    aPre[(long)bi * NH + col] = acc[v] + b2[col] * S[bi * NHEADS + h];
  }
}

// ---------------------------------------------------------------------------
// K8: LayerNorm over H per batch row. grid NB, block 256.
// ---------------------------------------------------------------------------
__global__ void k_ln(const float* __restrict__ aPre, const float* __restrict__ g,
                     const float* __restrict__ beta, float* __restrict__ out) {
  const int b = blockIdx.x, t = threadIdx.x;
  const float x0 = aPre[(long)b * NH + t];
  const float x1 = aPre[(long)b * NH + t + 256];
  float s = x0 + x1, s2 = x0 * x0 + x1 * x1;
  for (int o = 16; o > 0; o >>= 1) {
    s  += __shfl_xor(s, o, 32);
    s2 += __shfl_xor(s2, o, 32);
  }
  __shared__ float rs[8], rs2[8];
  __shared__ float muS, rstdS;
  const int w = t >> 5, lane = t & 31;
  if (lane == 0) { rs[w] = s; rs2[w] = s2; }
  __syncthreads();
  if (t == 0) {
    float S1 = 0.f, S2 = 0.f;
    for (int i = 0; i < 8; ++i) { S1 += rs[i]; S2 += rs2[i]; }
    float mu = S1 / NH;
    float var = S2 / NH - mu * mu;
    muS = mu;
    rstdS = rsqrtf(var + 1e-8f);
  }
  __syncthreads();
  const float mu = muS, r = rstdS;
  out[(long)b * NH + t]       = (x0 - mu) * r * g[t] + beta[t];
  out[(long)b * NH + t + 256] = (x1 - mu) * r * g[t + 256] + beta[t + 256];
}

// ---------------------------------------------------------------------------
extern "C" void kernel_launch(void* const* d_in, const int* in_sizes, int n_in,
                              void* d_out, int out_size, void* d_ws, size_t ws_size,
                              hipStream_t stream) {
  const float* ht1    = (const float*)d_in[0];
  const float* hidden = (const float*)d_in[1];
  const int*   mask   = (const int*)d_in[2];
  const float* W0 = (const float*)d_in[3];
  const float* b0 = (const float*)d_in[4];
  const float* W1 = (const float*)d_in[5];
  const float* b1 = (const float*)d_in[6];
  const float* W2 = (const float*)d_in[7];
  const float* b2 = (const float*)d_in[8];
  const float* lng = (const float*)d_in[9];
  const float* lnb = (const float*)d_in[10];

  float* out_a     = (float*)d_out;             // [B, H]
  float* out_alpha = out_a + (size_t)NB * NH;   // [B, L, HEADS]

  // Workspace carve-out (~19.3 MB total), all chunks 16B-aligned.
  float* q0   = (float*)d_ws;                         // B*H
  float* cbh  = q0 + (size_t)NB * NH;                 // B*HEADS
  float* Sbh  = cbh + (size_t)NB * NHEADS;            // B*HEADS
  float* aPre = Sbh + (size_t)NB * NHEADS;            // B*H
  float* sig  = aPre + (size_t)NB * NH;               // B*L*HEADS
  float* G    = sig + (size_t)NB * NL * NHEADS;       // B*HEADS*H
  _Float16* u16 = (_Float16*)(G + (size_t)NB * NHEADS * NH);  // B*HEADS*H halfs
  _Float16* wgt = u16 + (size_t)NB * NHEADS * NH;             // B*HEADS*NLP halfs

  k_q0     <<<dim3(32, 32),    32,  0, stream>>>(ht1, W0, b0, q0);
  k_c      <<<NB,              256, 0, stream>>>(q0, b1, cbh);
  k_u      <<<dim3(32, 32, 8), 32,  0, stream>>>(q0, W1, u16);
  k_scores <<<NB,              416, 0, stream>>>(u16, hidden, cbh, sig);
  k_softmax<<<NB,              256, 0, stream>>>(sig, mask, out_alpha, wgt, Sbh);
  k_wsum   <<<dim3(4, NB),     256, 0, stream>>>(wgt, hidden, G);
  k_proj   <<<dim3(4, 32, 8),  32,  0, stream>>>(G, W2, b2, Sbh, aPre);
  k_ln     <<<NB,              256, 0, stream>>>(aPre, lng, lnb, out_a);
}
